// MultiBoxLoss_64699387347203
// MI455X (gfx1250) — matrix operand order repaired and checked by reading, not compiled, not run on hardware
//
#include <hip/hip_runtime.h>
#include <cstdint>
#include <cstddef>

#define NN 16
#define PP 24576
#define CC 80
#define GG 16

typedef __attribute__((ext_vector_type(16))) _Float16 v16h;
typedef __attribute__((ext_vector_type(8)))  float    v8f;

__device__ __forceinline__ float fast_rcp(float x) { return __builtin_amdgcn_rcpf(x); }
__device__ __forceinline__ float sigmoidf_(float x) { return fast_rcp(1.f + __expf(-x)); }

__device__ __forceinline__ float iou4_(float ax1, float ay1, float ax2, float ay2, float aa,
                                       const float* bb, float ba) {
  float ix1 = fmaxf(ax1, bb[0]);
  float iy1 = fmaxf(ay1, bb[1]);
  float ix2 = fminf(ax2, bb[2]);
  float iy2 = fminf(ay2, bb[3]);
  float iw = fmaxf(ix2 - ix1, 0.f);
  float ih = fmaxf(iy2 - iy1, 0.f);
  float inter = iw * ih;
  return inter * fast_rcp(aa + ba - inter);
}

__device__ __forceinline__ float bl1_(float d) {
  const float b = 19.085536923187668f;              // e^(GAMMA/ALPHA) - 1, GAMMA/ALPHA = 3
  if (d < 0.11f)
    return (0.5f / b) * (b * d + 1.f) * __logf(b * d * (1.f / 0.11f) + 1.f) - 0.5f * d;
  return 1.5f * d + (1.5f / b - 0.055f);
}

// ---------------- K0: zero rowmax + scalar accumulators ----------------
__global__ void k0_zero(float* rowmax_c, float* rowmax_r, float* scal) {
  int t = threadIdx.x;
  if (t < NN * GG) { rowmax_c[t] = 0.f; rowmax_r[t] = 0.f; }
  if (t < 3) scal[t] = 0.f;
}

// ---------------- K1: qual_c/qual_r + WMMA one-hot cls_prob + rowmax ----------------
// grid (PP/128, NN), 256 threads (8 waves, each wave owns a 16-prior tile)
__global__ void __launch_bounds__(256) k1_qual(const float* __restrict__ loc,
                                               const float* __restrict__ conf,
                                               const float* __restrict__ priors,
                                               const float* __restrict__ targets,
                                               float* __restrict__ qual_c,
                                               float* __restrict__ qual_r,
                                               float* rowmax_c, float* rowmax_r) {
  __shared__ float s_pf[128][4];
  __shared__ float s_pfa[128];
  __shared__ float s_dec[128][4];
  __shared__ float s_deca[128];
  __shared__ float s_tr[GG][4];
  __shared__ float s_tra[GG];
  __shared__ int   s_lab[GG];

  const int n = blockIdx.y;
  const int pblk = blockIdx.x * 128;
  const int t = threadIdx.x;

  if (t < 128) {
    int p = pblk + t;
    float cx = priors[p * 4 + 0], cy = priors[p * 4 + 1];
    float pw = priors[p * 4 + 2], ph = priors[p * 4 + 3];
    s_pf[t][0] = cx - pw * 0.5f; s_pf[t][1] = cy - ph * 0.5f;
    s_pf[t][2] = cx + pw * 0.5f; s_pf[t][3] = cy + ph * 0.5f;
    s_pfa[t] = pw * ph;
    const float* lp = loc + ((size_t)n * PP + p) * 4;
    float bx = cx + lp[0] * 0.1f * pw;
    float by = cy + lp[1] * 0.1f * ph;
    float bw = pw * __expf(lp[2] * 0.2f);
    float bh = ph * __expf(lp[3] * 0.2f);
    s_dec[t][0] = bx - bw * 0.5f; s_dec[t][1] = by - bh * 0.5f;
    s_dec[t][2] = bx + bw * 0.5f; s_dec[t][3] = by + bh * 0.5f;
    s_deca[t] = bw * bh;
  } else if (t < 128 + GG) {
    int g = t - 128;
    const float* tp = targets + ((size_t)n * GG + g) * 5;
    s_tr[g][0] = tp[0]; s_tr[g][1] = tp[1]; s_tr[g][2] = tp[2]; s_tr[g][3] = tp[3];
    s_tra[g] = (tp[2] - tp[0]) * (tp[3] - tp[1]);
    s_lab[g] = (int)tp[4];
  }
  __syncthreads();

  const int w  = t >> 5;
  const int L  = t & 31;
  const int m  = L & 15;     // A row within tile / B column (= gt index g)
  const int hi = L >> 4;
  const int prow = pblk + w * 16 + m;
  const float* cp = conf + ((size_t)n * PP + prow) * CC;

  v8f acc = {};
#pragma unroll
  for (int kst = 0; kst < 3; ++kst) {
    const int k0 = kst * 32;
    v16h a, b;
    const int cA = k0 + hi * 8;
#pragma unroll
    for (int j = 0; j < 8; ++j) {
      int c1 = cA + j;
      a[j] = (c1 < CC) ? (_Float16)sigmoidf_(cp[c1]) : (_Float16)0.f;
      int c2 = cA + 16 + j;
      a[8 + j] = (c2 < CC) ? (_Float16)sigmoidf_(cp[c2]) : (_Float16)0.f;
    }
    const int lab = s_lab[m];
#pragma unroll
    for (int h = 0; h < 16; ++h) {
      int c = k0 + hi * 8 + (h < 8 ? h : h + 8);
      b[h] = (lab == c) ? (_Float16)1.f : (_Float16)0.f;
    }
    acc = __builtin_amdgcn_wmma_f32_16x16x32_f16(false, a, false, b, (short)0, acc, false, false);
  }

  // lane L holds cls_prob for g = m, priors m2 = v + hi*8 of this wave's tile
  const int g = m;
  const float t0 = s_tr[g][0], t1 = s_tr[g][1], t2 = s_tr[g][2], t3 = s_tr[g][3];
  const float ta = s_tra[g];
  float mc = 0.f, mr = 0.f;
#pragma unroll
  for (int v = 0; v < 8; ++v) {
    int m2 = v + hi * 8;
    int li = w * 16 + m2;
    int p  = pblk + w * 16 + m2;
    float acr = iou4_(t0, t1, t2, t3, ta, s_pf[li], s_pfa[li]);
    float reg = iou4_(t0, t1, t2, t3, ta, s_dec[li], s_deca[li]);
    float cls = acc[v];
    float qc = acr * __expf(reg * 0.5f);   // SIGMA = 2
    float qr = acr * __expf(cls * 0.5f);
    size_t o = ((size_t)n * GG + g) * PP + p;
    qual_c[o] = qc;
    qual_r[o] = qr;
    mc = fmaxf(mc, qc);
    mr = fmaxf(mr, qr);
  }
  atomicMax((int*)&rowmax_c[n * GG + g], __float_as_int(mc));
  atomicMax((int*)&rowmax_r[n * GG + g], __float_as_int(mr));
}

// ---------------- K2: per-(n,p) column max of normalized qual ----------------
__global__ void __launch_bounds__(256) k2_colmax(const float* __restrict__ qual_c,
                                                 const float* __restrict__ qual_r,
                                                 const float* rowmax_c, const float* rowmax_r,
                                                 float* colmax_c, float* colmax_r) {
  const int i = blockIdx.x * 256 + threadIdx.x;   // [0, NN*PP)
  const int n = i / PP;
  const int p = i - n * PP;
  __shared__ float invc[GG], invr[GG];
  if (threadIdx.x < GG) invc[threadIdx.x] = 1.f / fmaxf(rowmax_c[n * GG + threadIdx.x], 1e-6f);
  else if (threadIdx.x < 2 * GG) {
    int g = threadIdx.x - GG;
    invr[g] = 1.f / fmaxf(rowmax_r[n * GG + g], 1e-6f);
  }
  __syncthreads();
  float mc = 0.f, mr = 0.f;
#pragma unroll
  for (int g = 0; g < GG; ++g) {
    size_t o = ((size_t)n * GG + g) * PP + p;
    mc = fmaxf(mc, qual_c[o] * invc[g]);
    mr = fmaxf(mr, qual_r[o] * invr[g]);
  }
  colmax_c[i] = mc;
  colmax_r[i] = mr;
}

// ---------------- K3: exact 15th-largest per row (masked) ----------------
// grid 2*NN*GG blocks; block b: sel = b>>8, n = (b>>4)&15, g = b&15
__global__ void __launch_bounds__(256) k3_kth(const float* __restrict__ qual_c,
                                              const float* __restrict__ qual_r,
                                              const float* rowmax_c, const float* rowmax_r,
                                              const float* colmax_c, const float* colmax_r,
                                              float* kth_c, float* kth_r) {
  const int b = blockIdx.x;
  const int sel = b >> 8;
  const int n = (b >> 4) & 15;
  const int g = b & 15;
  const float* q  = sel ? qual_r : qual_c;
  const float* rm = sel ? rowmax_r : rowmax_c;
  const float* cm = sel ? colmax_r : colmax_c;
  const float inv = 1.f / fmaxf(rm[n * GG + g], 1e-6f);
  const float* qrow = q + ((size_t)n * GG + g) * PP;
  const float* cmn  = cm + (size_t)n * PP;
  const int tid = threadIdx.x;

  float t15[15];
#pragma unroll
  for (int k = 0; k < 15; ++k) t15[k] = 0.f;

  for (int p = tid; p < PP; p += 256) {
    float v = qrow[p] * inv;
    float mk = (v == cmn[p]) ? v : 0.f;
    if (mk > t15[14]) {
      t15[14] = mk;
#pragma unroll
      for (int k = 14; k > 0; --k) {
        if (t15[k] > t15[k - 1]) { float tmp = t15[k - 1]; t15[k - 1] = t15[k]; t15[k] = tmp; }
      }
    }
  }

  __shared__ float cand[256 * 15];
  __shared__ float rv[256];
  __shared__ int   ri[256];
#pragma unroll
  for (int k = 0; k < 15; ++k) cand[tid * 15 + k] = t15[k];
  __syncthreads();

  for (int it = 0; it < 15; ++it) {
    float bv = -2.f; int bi = -1;
#pragma unroll
    for (int k = 0; k < 15; ++k) {
      float v = cand[tid * 15 + k];
      if (v > bv) { bv = v; bi = tid * 15 + k; }
    }
    rv[tid] = bv; ri[tid] = bi;
    __syncthreads();
    for (int s = 128; s > 0; s >>= 1) {
      if (tid < s) {
        if (rv[tid + s] > rv[tid] || (rv[tid + s] == rv[tid] && ri[tid + s] < ri[tid])) {
          rv[tid] = rv[tid + s]; ri[tid] = ri[tid + s];
        }
      }
      __syncthreads();
    }
    float wval = rv[0];
    int widx = ri[0];
    if (widx >= tid * 15 && widx < tid * 15 + 15) cand[widx] = -1.f;
    if (it == 14 && tid == 0) {
      (sel ? kth_r : kth_c)[n * GG + g] = fmaxf(wval, 0.f);
    }
    __syncthreads();
  }
}

// ---------------- K4: per-prior assignment ----------------
__global__ void __launch_bounds__(256) k4_assign(const float* __restrict__ qual_c,
                                                 const float* __restrict__ qual_r,
                                                 const float* rowmax_c, const float* rowmax_r,
                                                 const float* __restrict__ colmax_c,
                                                 const float* __restrict__ colmax_r,
                                                 const float* kth_c, const float* kth_r,
                                                 const float* __restrict__ targets,
                                                 int* conf_t, float* score, float* wqr, int* gr,
                                                 float* scal) {
  const int i = blockIdx.x * 256 + threadIdx.x;
  const int n = i / PP;
  const int p = i - n * PP;
  __shared__ float invc[GG], invr[GG], kc[GG], kr[GG];
  __shared__ int lab[GG];
  if (threadIdx.x < GG) {
    int g = threadIdx.x;
    invc[g] = 1.f / fmaxf(rowmax_c[n * GG + g], 1e-6f);
    kc[g] = kth_c[n * GG + g];
    lab[g] = (int)targets[((size_t)n * GG + g) * 5 + 4];
  } else if (threadIdx.x < 2 * GG) {
    int g = threadIdx.x - GG;
    invr[g] = 1.f / fmaxf(rowmax_r[n * GG + g], 1e-6f);
    kr[g] = kth_r[n * GG + g];
  }
  __syncthreads();

  const float cmc = colmax_c[i], cmr = colmax_r[i];
  float qc = 0.f, qr = 0.f;
  int gc = 0, grv = 0;
#pragma unroll
  for (int g = 0; g < GG; ++g) {
    size_t o = ((size_t)n * GG + g) * PP + p;
    float vc = qual_c[o] * invc[g];
    float mkc = (vc == cmc) ? vc : 0.f;
    mkc = (mkc >= kc[g] && mkc > 0.f) ? mkc : 0.f;
    if (mkc > qc) { qc = mkc; gc = g; }
    float vr = qual_r[o] * invr[g];
    float mkr = (vr == cmr) ? vr : 0.f;
    mkr = (mkr >= kr[g] && mkr > 0.f) ? mkr : 0.f;
    if (mkr > qr) { qr = mkr; grv = g; }
  }
  conf_t[i] = (qc > 0.f) ? (lab[gc] + 1) : 0;
  score[i] = qc;          // = relu(overlap_t - 3) since IoU <= 1
  wqr[i] = qr;            // = relu(pred_t - 3)
  gr[i] = grv;

  __shared__ float red[256];
  red[threadIdx.x] = qr;
  __syncthreads();
  for (int s = 128; s > 0; s >>= 1) {
    if (threadIdx.x < s) red[threadIdx.x] += red[threadIdx.x + s];
    __syncthreads();
  }
  if (threadIdx.x == 0) atomicAdd(&scal[0], red[0]);
}

// ---------------- K5: weighted balanced-L1 localization loss (numerator) ----------------
__global__ void __launch_bounds__(256) k5_locloss(const float* __restrict__ loc,
                                                  const float* __restrict__ priors,
                                                  const float* __restrict__ targets,
                                                  const float* __restrict__ wqr,
                                                  const int* __restrict__ gr,
                                                  float* scal) {
  const int i = blockIdx.x * 256 + threadIdx.x;
  const int n = i / PP;
  const int p = i - n * PP;
  float acc = 0.f;
  float w = wqr[i];
  if (w > 0.f) {
    int g = gr[i];
    const float* tp = targets + ((size_t)n * GG + g) * 5;
    float x1 = tp[0], y1 = tp[1], x2 = tp[2], y2 = tp[3];
    float cx = priors[p * 4 + 0], cy = priors[p * 4 + 1];
    float pw = priors[p * 4 + 2], ph = priors[p * 4 + 3];
    float e0 = ((x1 + x2) * 0.5f - cx) / (0.1f * pw);
    float e1 = ((y1 + y2) * 0.5f - cy) / (0.1f * ph);
    float e2 = __logf(fmaxf((x2 - x1) / pw, 1e-6f)) * 5.f;   // / VAR1 (0.2)
    float e3 = __logf(fmaxf((y2 - y1) / ph, 1e-6f)) * 5.f;
    const float* lp = loc + ((size_t)n * PP + p) * 4;
    acc = w * (bl1_(fabsf(lp[0] - e0)) + bl1_(fabsf(lp[1] - e1)) +
               bl1_(fabsf(lp[2] - e2)) + bl1_(fabsf(lp[3] - e3)));
  }
  __shared__ float red[256];
  red[threadIdx.x] = acc;
  __syncthreads();
  for (int s = 128; s > 0; s >>= 1) {
    if (threadIdx.x < s) red[threadIdx.x] += red[threadIdx.x + s];
    __syncthreads();
  }
  if (threadIdx.x == 0) atomicAdd(&scal[1], red[0]);
}

// ---------------- K6: conf loss, async global->LDS double-buffered stream ----------------
// grid 1920 x 256 => stride 491520 vec4; total vec4 = 7864320 => exactly 16 uniform iters
__global__ void __launch_bounds__(256) k6_confloss(const float* __restrict__ conf,
                                                   const int* __restrict__ conf_t,
                                                   const float* __restrict__ score,
                                                   float* scal) {
  __shared__ float4 stage[2][256];
  const int tid = threadIdx.x;
  const size_t stride = (size_t)1920 * 256;
  size_t idx = (size_t)blockIdx.x * 256 + tid;

  unsigned lds0 = (unsigned)(size_t)(void*)&stage[0][tid];
  unsigned lds1 = (unsigned)(size_t)(void*)&stage[1][tid];

  // prologue: stage iteration 0
  asm volatile("global_load_async_to_lds_b128 %0, %1, off"
               :: "v"(lds0), "v"((unsigned long long)(size_t)(conf + idx * 4))
               : "memory");

  float acc = 0.f;
  int cur = 0;
  for (int it = 0; it < 16; ++it) {
    if (it < 15) {
      unsigned ldsn = (cur == 0) ? lds1 : lds0;
      asm volatile("global_load_async_to_lds_b128 %0, %1, off"
                   :: "v"(ldsn), "v"((unsigned long long)(size_t)(conf + (idx + stride) * 4))
                   : "memory");
      asm volatile("s_wait_asynccnt 0x1" ::: "memory");
    } else {
      asm volatile("s_wait_asynccnt 0x0" ::: "memory");
    }
    float4 v = stage[cur][tid];
    size_t row = idx / 20;                 // (n*PP + p),  C/4 = 20
    int c0 = (int)(idx % 20) * 4;
    int ct = conf_t[row];
    float sc = score[row];
    const float* vf = &v.x;
#pragma unroll
    for (int k = 0; k < 4; ++k) {
      float x = vf[k];
      float tgt = (ct == (c0 + k + 1)) ? sc : 0.f;
      float bce = fmaxf(x, 0.f) - x * tgt + log1pf(__expf(-fabsf(x)));
      float sg = sigmoidf_(x);
      float d = sg - tgt;
      acc += bce * d * d;
    }
    idx += stride;
    cur ^= 1;
  }

  __shared__ float red[256];
  red[tid] = acc;
  __syncthreads();
  for (int s = 128; s > 0; s >>= 1) {
    if (tid < s) red[tid] += red[tid + s];
    __syncthreads();
  }
  if (tid == 0) atomicAdd(&scal[2], red[0]);
}

// ---------------- K7: finalize ----------------
__global__ void k7_final(const float* scal, float* out) {
  // loss_l = sum(w * bl1) / (4 * sum_q_r)  (w broadcast over 4 coords, normalized)
  out[0] = scal[1] / fmaxf(4.f * scal[0], 1e-20f);
  out[1] = scal[2];
}

extern "C" void kernel_launch(void* const* d_in, const int* in_sizes, int n_in,
                              void* d_out, int out_size, void* d_ws, size_t ws_size,
                              hipStream_t stream) {
  const float* loc     = (const float*)d_in[0];
  const float* conf    = (const float*)d_in[1];
  const float* priors  = (const float*)d_in[2];
  const float* targets = (const float*)d_in[3];
  float* out = (float*)d_out;

  char* w = (char*)d_ws;
  const size_t NP  = (size_t)NN * PP;        // 393216
  const size_t NGP = NP * GG;                // 6291456
  float* qual_c   = (float*)w; w += NGP * 4;
  float* qual_r   = (float*)w; w += NGP * 4;
  float* colmax_c = (float*)w; w += NP * 4;
  float* colmax_r = (float*)w; w += NP * 4;
  float* score    = (float*)w; w += NP * 4;
  float* wqr      = (float*)w; w += NP * 4;
  int*   conf_t   = (int*)w;   w += NP * 4;
  int*   gr       = (int*)w;   w += NP * 4;
  float* rowmax_c = (float*)w; w += 256 * 4;
  float* rowmax_r = (float*)w; w += 256 * 4;
  float* kth_c    = (float*)w; w += 256 * 4;
  float* kth_r    = (float*)w; w += 256 * 4;
  float* scal     = (float*)w; w += 256;

  k0_zero<<<1, 256, 0, stream>>>(rowmax_c, rowmax_r, scal);
  k1_qual<<<dim3(PP / 128, NN), 256, 0, stream>>>(loc, conf, priors, targets,
                                                  qual_c, qual_r, rowmax_c, rowmax_r);
  k2_colmax<<<(NN * PP) / 256, 256, 0, stream>>>(qual_c, qual_r, rowmax_c, rowmax_r,
                                                 colmax_c, colmax_r);
  k3_kth<<<2 * NN * GG, 256, 0, stream>>>(qual_c, qual_r, rowmax_c, rowmax_r,
                                          colmax_c, colmax_r, kth_c, kth_r);
  k4_assign<<<(NN * PP) / 256, 256, 0, stream>>>(qual_c, qual_r, rowmax_c, rowmax_r,
                                                 colmax_c, colmax_r, kth_c, kth_r, targets,
                                                 conf_t, score, wqr, gr, scal);
  k5_locloss<<<(NN * PP) / 256, 256, 0, stream>>>(loc, priors, targets, wqr, gr, scal);
  k6_confloss<<<1920, 256, 0, stream>>>(conf, conf_t, score, scal);
  k7_final<<<1, 1, 0, stream>>>(scal, out);
}
